// MyModel_59365037965463
// MI455X (gfx1250) — matrix operand order repaired and checked by reading, not compiled, not run on hardware
//
#include <hip/hip_runtime.h>
#include <hip/hip_bf16.h>

#define HS   128
#define FIN  64

typedef __attribute__((ext_vector_type(16))) _Float16 v16h;
typedef __attribute__((ext_vector_type(8)))  _Float16 h8;
typedef __attribute__((ext_vector_type(8)))  float    v8f;

// ---------------------------------------------------------------- WMMA utils

__device__ __forceinline__ v8f wmma_f16(v16h a, v16h b, v8f c) {
    // D = A(16x32 f16) * B(32x16 f16) + C(16x16 f32)
    return __builtin_amdgcn_wmma_f32_16x16x32_f16(
        /*neg_a=*/false, a, /*neg_b=*/false, b,
        /*c_mod=*/(short)0, c, /*reuse_a=*/false, /*reuse_b=*/false);
}

// B tiles are pre-packed per-lane: pk[((kc*8+nt)*32+lane)*16 + e]
__device__ __forceinline__ v16h load_b_tile(const _Float16* pk, int kc, int nt, int lane) {
    return *(const v16h*)(pk + ((((kc << 3) + nt) * 32 + lane) << 4));
}

// A operand, f16 row-major source. ISA 16-bit A layout:
//   half e in [0,8)  -> K = k0 + 8*hi + e
//   half e in [8,16) -> K = k0 + 16 + 8*hi + (e-8)
__device__ __forceinline__ v16h load_a_f16(const _Float16* rowp, int k0, int hi) {
    h8 lo = *(const h8*)(rowp + k0 + 8 * hi);
    h8 hh = *(const h8*)(rowp + k0 + 16 + 8 * hi);
    v16h a;
#pragma unroll
    for (int e = 0; e < 8; ++e) { a[e] = lo[e]; a[8 + e] = hh[e]; }
    return a;
}

// A operand from f32 row-major source, scaled by s, converted to f16.
__device__ __forceinline__ v16h load_a_f32s(const float* rowp, int k0, int hi, float s) {
    const float4 u0 = *(const float4*)(rowp + k0 + 8 * hi);
    const float4 u1 = *(const float4*)(rowp + k0 + 8 * hi + 4);
    const float4 w0 = *(const float4*)(rowp + k0 + 16 + 8 * hi);
    const float4 w1 = *(const float4*)(rowp + k0 + 16 + 8 * hi + 4);
    float f[16] = {u0.x, u0.y, u0.z, u0.w, u1.x, u1.y, u1.z, u1.w,
                   w0.x, w0.y, w0.z, w0.w, w1.x, w1.y, w1.z, w1.w};
    v16h a;
#pragma unroll
    for (int e = 0; e < 16; ++e) a[e] = (_Float16)(f[e] * s);
    return a;
}

// ------------------------------------------------------------ weight repack
// Packs W1(eff, K=64), Wl(K=128), Wr(K=128), W2(K=128) into WMMA-B layout:
//   element e of lane -> K = kc*32 + 8*(lane/16) + e (e<8) or kc*32 + 16 + 8*(lane/16) + (e-8) (e>=8), N = nt*16 + (lane%16)
// slots: 0-1 = W1 (kc=slot), 2-5 = Wl, 6-9 = Wr, 10-13 = W2.
__global__ void repack_kernel(const float* __restrict__ W1, const float* __restrict__ Wl,
                              const float* __restrict__ Wr, const float* __restrict__ W2,
                              _Float16* __restrict__ W1p, _Float16* __restrict__ Wlp,
                              _Float16* __restrict__ Wrp, _Float16* __restrict__ W2p) {
    int t = blockIdx.x * blockDim.x + threadIdx.x;
    if (t >= 14 * 8 * 32) return;
    int lane = t & 31;
    int nt   = (t >> 5) & 7;
    int slot = t >> 8;
    int n  = nt * 16 + (lane & 15);
    int hi = lane >> 4;

    const float* src; _Float16* dst; int kc; int which;
    if (slot < 2)       { which = 0; kc = slot;      src = W1; dst = W1p; }
    else if (slot < 6)  { which = 1; kc = slot - 2;  src = Wl; dst = Wlp; }
    else if (slot < 10) { which = 2; kc = slot - 6;  src = Wr; dst = Wrp; }
    else                { which = 3; kc = slot - 10; src = W2; dst = W2p; }

    _Float16* out = dst + (((kc * 8 + nt) * 32 + lane) << 4);
#pragma unroll
    for (int e = 0; e < 16; ++e) {
        int k = kc * 32 + (e < 8 ? 8 * hi + e : 16 + 8 * hi + (e - 8));
        float v;
        if (which == 0) {            // Weff[k][n]: price row k==0 is zero
            v = (k == 0) ? 0.0f : src[(size_t)n * 63 + (k - 1)];
        } else if (which == 3) {     // W2T[k][n] = W2[n][k]
            v = src[(size_t)n * 128 + k];
        } else {                     // Wl/Wr: [128 x 129], first 128 cols
            v = src[(size_t)n * 129 + k];
        }
        out[e] = (_Float16)v;
    }
}

// ------------------------------------------------------ degree & price mean
__global__ void edge_price_kernel(const int* __restrict__ ei, const float* __restrict__ x,
                                  float* __restrict__ cnt, float* __restrict__ psum, int E) {
    int e = blockIdx.x * blockDim.x + threadIdx.x;
    if (e >= E) return;
    int s = ei[e];
    int d = ei[(size_t)E + e];
    atomicAdd(cnt + d, 1.0f);
    atomicAdd(psum + d, x[(size_t)s * FIN]);
}

__global__ void pmean_kernel(const float* __restrict__ psum, const float* __restrict__ cnt,
                             float* __restrict__ pmean, float* __restrict__ rinv, int N) {
    int i = blockIdx.x * blockDim.x + threadIdx.x;
    if (i >= N) return;
    float inv = 1.0f / fmaxf(cnt[i], 1.0f);
    pmean[i] = psum[i] * inv;
    rinv[i]  = inv;
}

// ---------------------------------------------------------------- fc1 + tanh
// One wave (32 threads) per 16-node tile. h16[N][128] = tanh(x @ Weff + b1).
__global__ __launch_bounds__(32) void fc1_kernel(const float* __restrict__ x,
                                                 const float* __restrict__ b1,
                                                 const _Float16* __restrict__ W1p,
                                                 _Float16* __restrict__ h16) {
    __shared__ __align__(32) _Float16 stage[16 * 128];
    const int tile = blockIdx.x;
    const int lane = threadIdx.x;
    const int n  = lane & 15;
    const int hi = lane >> 4;
    const int row = tile * 16 + n;

    const float* xr = x + (size_t)row * FIN;
    v16h a0 = load_a_f32s(xr, 0, hi, 1.0f);
    v16h a1 = load_a_f32s(xr, 32, hi, 1.0f);

#pragma unroll
    for (int nt = 0; nt < 8; ++nt) {
        float bv = b1[nt * 16 + n];
        v8f c;
#pragma unroll
        for (int r = 0; r < 8; ++r) c[r] = bv;
        c = wmma_f16(a0, load_b_tile(W1p, 0, nt, lane), c);
        c = wmma_f16(a1, load_b_tile(W1p, 1, nt, lane), c);
#pragma unroll
        for (int r = 0; r < 8; ++r)
            stage[(8 * hi + r) * 128 + nt * 16 + n] = (_Float16)tanhf(c[r]);
    }
    __syncthreads();
    // coalesced row-major write: lane handles row (lane&15), half hi of 128 cols
    _Float16* dstp = h16 + (size_t)(tile * 16 + n) * 128 + hi * 64;
    const _Float16* srcp = stage + n * 128 + hi * 64;
#pragma unroll
    for (int i = 0; i < 8; ++i)
        *(h8*)(dstp + i * 8) = *(const h8*)(srcp + i * 8);
}

// ------------------------------------------------------------- edge scatter
// 8 lanes per edge; f16 gather of h[src] (L2-resident), f32 atomic accumulate.
__global__ void scatter_kernel(const int* __restrict__ ei, const _Float16* __restrict__ h16,
                               const float* __restrict__ pmean,
                               float* __restrict__ acc, float* __restrict__ accp, int E) {
    long long t = (long long)blockIdx.x * blockDim.x + threadIdx.x;
    int e = (int)(t >> 3);
    int j = (int)(t & 7);
    if (e >= E) return;
    int s = ei[e];
    int d = ei[(size_t)E + e];
    const h8 v0 = *(const h8*)(h16 + (size_t)s * 128 + j * 16);
    const h8 v1 = *(const h8*)(h16 + (size_t)s * 128 + j * 16 + 8);
    float* ad = acc + (size_t)d * 128 + j * 16;
#pragma unroll
    for (int i = 0; i < 8; ++i) atomicAdd(ad + i, (float)v0[i]);
#pragma unroll
    for (int i = 0; i < 8; ++i) atomicAdd(ad + 8 + i, (float)v1[i]);
    if (j == 0) atomicAdd(accp + d, pmean[s]);
}

// ---------------------------------------------- fused SAGE + fc2 + fc3 head
// One wave per 16-node tile:
//  out1 = agg@WlT + h@WrT + bl + aggp⊗Wl[:,128] + pmean⊗Wr[:,128]
//  out2 = relu(out1@W2T + b2) ;  y = out2@w3 + b3
__global__ __launch_bounds__(32) void sage_kernel(
    const _Float16* __restrict__ h16, const float* __restrict__ acc,
    const float* __restrict__ accp, const float* __restrict__ pmean,
    const float* __restrict__ rinv,
    const float* __restrict__ Wl, const float* __restrict__ bl,
    const float* __restrict__ Wr, const float* __restrict__ b2,
    const float* __restrict__ W3, const float* __restrict__ b3,
    const _Float16* __restrict__ Wlp, const _Float16* __restrict__ Wrp,
    const _Float16* __restrict__ W2p, float* __restrict__ y) {
    __shared__ __align__(32) _Float16 stage[16 * 128];
    __shared__ float s_aggp[16];
    __shared__ float s_pm[16];

    const int tile = blockIdx.x;
    const int lane = threadIdx.x;
    const int n  = lane & 15;
    const int hi = lane >> 4;
    const int row = tile * 16 + n;

    if (lane < 16) {
        int rr = tile * 16 + lane;
        float rv = rinv[rr];
        s_aggp[lane] = accp[rr] * rv;
        s_pm[lane]   = pmean[rr];
    }
    __syncthreads();

    const float rv = rinv[row];
    const _Float16* hr = h16 + (size_t)row * 128;
    const float*    ar = acc + (size_t)row * 128;
    v16h ah[4], ag[4];
#pragma unroll
    for (int kc = 0; kc < 4; ++kc) {
        ah[kc] = load_a_f16(hr, kc * 32, hi);
        ag[kc] = load_a_f32s(ar, kc * 32, hi, rv);
    }

#pragma unroll
    for (int nt = 0; nt < 8; ++nt) {
        const int ng = nt * 16 + n;
        const float c0  = bl[ng];
        const float wll = Wl[(size_t)ng * 129 + 128];
        const float wrl = Wr[(size_t)ng * 129 + 128];
        v8f c;
#pragma unroll
        for (int r = 0; r < 8; ++r)
            c[r] = c0 + s_aggp[8 * hi + r] * wll + s_pm[8 * hi + r] * wrl;
#pragma unroll
        for (int kc = 0; kc < 4; ++kc)
            c = wmma_f16(ag[kc], load_b_tile(Wlp, kc, nt, lane), c);
#pragma unroll
        for (int kc = 0; kc < 4; ++kc)
            c = wmma_f16(ah[kc], load_b_tile(Wrp, kc, nt, lane), c);
#pragma unroll
        for (int r = 0; r < 8; ++r)
            stage[(8 * hi + r) * 128 + ng] = (_Float16)c[r];
    }
    __syncthreads();

    v16h a2[4];
#pragma unroll
    for (int kc = 0; kc < 4; ++kc)
        a2[kc] = load_a_f16(stage + n * 128, kc * 32, hi);
    __syncthreads();

#pragma unroll
    for (int nt = 0; nt < 8; ++nt) {
        const int ng = nt * 16 + n;
        float bv = b2[ng];
        v8f c;
#pragma unroll
        for (int r = 0; r < 8; ++r) c[r] = bv;
#pragma unroll
        for (int kc = 0; kc < 4; ++kc)
            c = wmma_f16(a2[kc], load_b_tile(W2p, kc, nt, lane), c);
#pragma unroll
        for (int r = 0; r < 8; ++r) {
            float vv = c[r];
            stage[(8 * hi + r) * 128 + ng] = (_Float16)(vv > 0.0f ? vv : 0.0f);
        }
    }
    __syncthreads();

    if (lane < 16) {
        float a3 = b3[0];
        const _Float16* sr = stage + lane * 128;
#pragma unroll 16
        for (int k = 0; k < 128; ++k) a3 += (float)sr[k] * W3[k];
        y[tile * 16 + lane] = a3;
    }
}

// -------------------------------------------------------------------- launch

static inline size_t align256(size_t x) { return (x + 255) & ~(size_t)255; }

extern "C" void kernel_launch(void* const* d_in, const int* in_sizes, int n_in,
                              void* d_out, int out_size, void* d_ws, size_t ws_size,
                              hipStream_t stream) {
    const float* x  = (const float*)d_in[0];
    const int*   ei = (const int*)  d_in[1];
    const float* W1 = (const float*)d_in[2];
    const float* b1 = (const float*)d_in[3];
    const float* Wl = (const float*)d_in[4];
    const float* bl = (const float*)d_in[5];
    const float* Wr = (const float*)d_in[6];
    const float* W2 = (const float*)d_in[7];
    const float* b2 = (const float*)d_in[8];
    const float* W3 = (const float*)d_in[9];
    const float* b3 = (const float*)d_in[10];

    const int N = in_sizes[0] / FIN;
    const int E = in_sizes[1] / 2;
    const int tiles = N / 16;   // N = 200000 = 12500 * 16

    // workspace layout
    char* ws = (char*)d_ws;
    size_t off = 0;
    float*    acc   = (float*)   (ws + off); off = align256(off + (size_t)N * 128 * 4);
    _Float16* h16   = (_Float16*)(ws + off); off = align256(off + (size_t)N * 128 * 2);
    float*    psum  = (float*)   (ws + off); off = align256(off + (size_t)N * 4);
    float*    cnt   = (float*)   (ws + off); off = align256(off + (size_t)N * 4);
    float*    pmean = (float*)   (ws + off); off = align256(off + (size_t)N * 4);
    float*    accp  = (float*)   (ws + off); off = align256(off + (size_t)N * 4);
    float*    rinv  = (float*)   (ws + off); off = align256(off + (size_t)N * 4);
    _Float16* W1p   = (_Float16*)(ws + off); off = align256(off + (size_t)2 * 8 * 32 * 16 * 2);
    _Float16* Wlp   = (_Float16*)(ws + off); off = align256(off + (size_t)4 * 8 * 32 * 16 * 2);
    _Float16* Wrp   = (_Float16*)(ws + off); off = align256(off + (size_t)4 * 8 * 32 * 16 * 2);
    _Float16* W2p   = (_Float16*)(ws + off); off = align256(off + (size_t)4 * 8 * 32 * 16 * 2);

    // zero the accumulators (graph-capturable)
    hipMemsetAsync(acc,  0, (size_t)N * 128 * 4, stream);
    hipMemsetAsync(psum, 0, (size_t)N * 4, stream);
    hipMemsetAsync(cnt,  0, (size_t)N * 4, stream);
    hipMemsetAsync(accp, 0, (size_t)N * 4, stream);

    repack_kernel<<<14, 256, 0, stream>>>(W1, Wl, Wr, W2, W1p, Wlp, Wrp, W2p);

    edge_price_kernel<<<(E + 255) / 256, 256, 0, stream>>>(ei, x, cnt, psum, E);
    pmean_kernel<<<(N + 255) / 256, 256, 0, stream>>>(psum, cnt, pmean, rinv, N);

    fc1_kernel<<<tiles, 32, 0, stream>>>(x, b1, W1p, h16);

    {
        long long th = (long long)E * 8;
        int blocks = (int)((th + 255) / 256);
        scatter_kernel<<<blocks, 256, 0, stream>>>(ei, h16, pmean, acc, accp, E);
    }

    sage_kernel<<<tiles, 32, 0, stream>>>(h16, acc, accp, pmean, rinv,
                                          Wl, bl, Wr, b2, W3, b3,
                                          Wlp, Wrp, W2p, (float*)d_out);

    // tuple output: append edge_index verbatim after the N floats
    hipMemcpyAsync((char*)d_out + (size_t)N * sizeof(float), (const void*)ei,
                   (size_t)2 * E * sizeof(int), hipMemcpyDeviceToDevice, stream);
}
